// AttModel_32487132627248
// MI455X (gfx1250) — matrix operand order repaired and checked by reading, not compiled, not run on hardware
//
#include <hip/hip_runtime.h>
#include <hip/hip_bf16.h>

typedef _Float16 h16;
typedef __attribute__((ext_vector_type(16))) _Float16 v16h;
typedef __attribute__((ext_vector_type(8)))  _Float16 v8h;
typedef __attribute__((ext_vector_type(8)))  float    v8f;

#define BSZ   256
#define INF   54
#define DDIM  256
#define KSW   10
#define CHN   20
#define INN   50
#define NSTG  12
#define VNW   31
#define NSAMP (BSZ*VNW)   // 7936

#define GF_ACC  1
#define GF_RELU 2
#define GF_BIAS 4
#define GF_F16  8

// ---------------------------------------------------------------------------
// Optional TDM probe (amdgpu-toolchain 6-arg builtin per bridge doc). Never
// launched; NULL descriptor (count=0). Compiled out if builtin absent.
// ---------------------------------------------------------------------------
#if __has_builtin(__builtin_amdgcn_tensor_load_to_lds)
typedef __attribute__((ext_vector_type(4))) unsigned tdm_u4;
typedef __attribute__((ext_vector_type(8))) int      tdm_i8;
typedef __attribute__((ext_vector_type(4))) int      tdm_i4;
__global__ void k_tdm_probe()
{
    tdm_u4 g0 = {0u,0u,0u,0u};
    tdm_i8 g1 = {0,0,0,0,0,0,0,0};
    tdm_i4 g2 = {0,0,0,0};
    tdm_i4 g3 = {0,0,0,0};
    tdm_i8 g4 = {0,0,0,0,0,0,0,0};
    __builtin_amdgcn_tensor_load_to_lds(g0, g1, g2, g3, g4, 0);
#if __has_builtin(__builtin_amdgcn_s_wait_tensorcnt)
    __builtin_amdgcn_s_wait_tensorcnt(0);
#endif
}
#endif

// ---------------------------------------------------------------------------
// FAST GEMM: C[M,N] (+=) A[M,K]*BT[N,K]^T, f16 in / f32 acc, WMMA 16x16x32.
// BT is the weight matrix stored n-major / k-contiguous (row stride ldb).
// Host-enforced: M%64==0, N%128==0, K%32==0, lda%8==0, ldb%8==0.
// Block = 256 thr = 8 waves (2x4); block tile 64x128; wave tile 32x32
// (4 v_wmma per K-step). All global staging, LDS staging AND fragment
// loads are b128 — zero scalar traffic in the hot loop.
// ---------------------------------------------------------------------------
__global__ __launch_bounds__(256)
void gemm_wmma_fast(const h16* __restrict__ A, const h16* __restrict__ B,
                    float* __restrict__ C, h16* __restrict__ Ch,
                    const float* __restrict__ bias,
                    int M, int N, int K, int lda, int ldb, int ldc,
                    int rowDiv, int rowMul, int rowAdd, int flags)
{
    __shared__ h16 As[64][48];    // 64 m x 32 k (+pad, 96B row stride)
    __shared__ h16 Bt[128][48];   // 128 n x 32 k (+pad, 96B row stride)

    const int tid  = threadIdx.x;
    const int lane = tid & 31;
    const int wave = tid >> 5;
    const int wmB  = (wave >> 2) * 32;   // 0 / 32
    const int wnB  = (wave & 3) * 32;    // 0..96
    const int bm0  = blockIdx.y * 64;
    const int bn0  = blockIdx.x * 128;

    // A staging: thread t -> row tid>>2, 8-half chunk (tid&3)*8
    const int ar  = tid >> 2;
    const int ac0 = (tid & 3) * 8;
    int mg = bm0 + ar;
    int arow = (rowDiv > 0) ? ((mg / rowDiv) * rowMul + (mg % rowDiv) + rowAdd) : mg;
    const h16* Arow = A + (long long)arow * lda;

    v8f acc[2][2];
    #pragma unroll
    for (int mi = 0; mi < 2; ++mi)
        #pragma unroll
        for (int ni = 0; ni < 2; ++ni)
            acc[mi][ni] = (v8f){0.f,0.f,0.f,0.f,0.f,0.f,0.f,0.f};

    for (int k0 = 0; k0 < K; k0 += 32) {
        // stage A (one b128 per thread)
        *(v8h*)&As[ar][ac0] = *(const v8h*)&Arow[k0 + ac0];
        // stage BT (two b128 per thread): rows n, k-contiguous
        #pragma unroll
        for (int it = 0; it < 2; ++it) {
            int e   = tid + it * 256;
            int bn  = e >> 2, bc0 = (e & 3) * 8;
            *(v8h*)&Bt[bn][bc0] =
                *(const v8h*)&B[(long long)(bn0 + bn) * ldb + k0 + bc0];
        }
        if (k0 + 32 < K) {
            __builtin_prefetch(&Arow[k0 + 32 + ac0], 0, 1);
            __builtin_prefetch(&B[(long long)(bn0 + (tid & 127)) * ldb + k0 + 32], 0, 1);
        }
        __syncthreads();

        // fragments per ISA 7.12.2 (wave32); A and B both use the table k order
        const int mn = lane & 15;
        const int g  = lane >> 4;
        v16h a0, a1, b0, b1;
        #pragma unroll
        for (int hh = 0; hh < 16; ++hh) {
            int kk = 8*g + (hh & 7) + ((hh >> 3) << 4);
            a0[hh] = As[wmB +      mn][kk];
            a1[hh] = As[wmB + 16 + mn][kk];
        }
        #pragma unroll
        for (int hh = 0; hh < 16; ++hh) {
            b0[hh] = Bt[wnB +      mn][8*g + (hh & 7) + ((hh >> 3) << 4)];
            b1[hh] = Bt[wnB + 16 + mn][8*g + (hh & 7) + ((hh >> 3) << 4)];
        }
        acc[0][0] = __builtin_amdgcn_wmma_f32_16x16x32_f16(false, a0, false, b0, (short)0, acc[0][0], false, false);
        acc[0][1] = __builtin_amdgcn_wmma_f32_16x16x32_f16(false, a0, false, b1, (short)0, acc[0][1], false, false);
        acc[1][0] = __builtin_amdgcn_wmma_f32_16x16x32_f16(false, a1, false, b0, (short)0, acc[1][0], false, false);
        acc[1][1] = __builtin_amdgcn_wmma_f32_16x16x32_f16(false, a1, false, b1, (short)0, acc[1][1], false, false);
        __syncthreads();
    }

    // epilogue: lane%16 -> N, VGPR v -> M = v + 8*(lane/16)
    const int nn = lane & 15;
    const int g2 = lane >> 4;
    #pragma unroll
    for (int mi = 0; mi < 2; ++mi) {
        #pragma unroll
        for (int ni = 0; ni < 2; ++ni) {
            int colG = bn0 + wnB + ni*16 + nn;
            float bv = (flags & GF_BIAS) ? bias[colG] : 0.f;
            #pragma unroll
            for (int vv = 0; vv < 8; ++vv) {
                int rowG = bm0 + wmB + mi*16 + vv + 8*g2;
                long long ci = (long long)rowG * ldc + colG;
                float x = acc[mi][ni][vv] + bv;
                if (flags & GF_ACC)  x += C[ci];
                if (flags & GF_RELU) x = fmaxf(x, 0.f);
                C[ci] = x;
                if ((flags & GF_F16) && Ch) Ch[ci] = (h16)x;
            }
        }
    }
}

// ---------------------------------------------------------------------------
// General (edge/batched) GEMM: B stored [k][n] row-major; predicated staging;
// any M/N/K, batch strides, A-row remap. Used only for ragged shapes.
// ---------------------------------------------------------------------------
__global__ __launch_bounds__(256)
void gemm_wmma(const h16* __restrict__ A, const h16* __restrict__ B,
               float* __restrict__ C, h16* __restrict__ Ch,
               const float* __restrict__ bias,
               int M, int N, int K, int lda, int ldb, int ldc,
               long long sA, long long sB, long long sC,
               int rowDiv, int rowMul, int rowAdd, int flags)
{
    __shared__ h16 As[32][40];
    __shared__ h16 Bs[32][72];

    const int tid  = threadIdx.x;
    const int lane = tid & 31;
    const int wave = tid >> 5;
    const int wm   = wave >> 2;
    const int wn   = wave & 3;
    const int bm0  = blockIdx.y * 32;
    const int bn0  = blockIdx.x * 64;
    const long long bz = blockIdx.z;

    A += bz * sA;  B += bz * sB;  C += bz * sC;
    if (Ch) Ch += bz * sC;

    v8f acc = {0.f,0.f,0.f,0.f,0.f,0.f,0.f,0.f};

    const int lr  = tid >> 3;
    const int lc0 = (tid & 7) * 4;
    long long aoff = -1;
    {
        int mg = bm0 + lr;
        if (mg < M) {
            int ar = (rowDiv > 0) ? ((mg / rowDiv) * rowMul + (mg % rowDiv) + rowAdd) : mg;
            aoff = (long long)ar * lda;
        }
    }

    for (int k0 = 0; k0 < K; k0 += 32) {
        #pragma unroll
        for (int u = 0; u < 4; ++u) {
            int kk = k0 + lc0 + u;
            h16 v = (h16)0.f;
            if (aoff >= 0 && kk < K) v = A[aoff + kk];
            As[lr][lc0 + u] = v;
        }
        #pragma unroll
        for (int u = 0; u < 8; ++u) {
            int e = tid * 8 + u;
            int r = e >> 6, c = e & 63;
            int kk = k0 + r, ng = bn0 + c;
            h16 v = (h16)0.f;
            if (kk < K && ng < N) v = B[(long long)kk * ldb + ng];
            Bs[r][c] = v;
        }
        __syncthreads();

        v16h af, bf;
        {
            const int mn = lane & 15;
            const int g  = lane >> 4;
            #pragma unroll
            for (int hh = 0; hh < 16; ++hh) {
                int kk = 8*g + (hh & 7) + ((hh >> 3) << 4);
                af[hh] = As[wm*16 + mn][kk];
            }
            #pragma unroll
            for (int hh = 0; hh < 16; ++hh)
                bf[hh] = Bs[8*g + (hh & 7) + ((hh >> 3) << 4)][wn*16 + mn];
        }
        acc = __builtin_amdgcn_wmma_f32_16x16x32_f16(false, af, false, bf,
                                                     (short)0, acc, false, false);
        __syncthreads();
    }

    const int nn   = lane & 15;
    const int g2   = lane >> 4;
    const int colG = bn0 + wn*16 + nn;
    if (colG < N) {
        float bv = (flags & GF_BIAS) ? bias[colG] : 0.f;
        #pragma unroll
        for (int vv = 0; vv < 8; ++vv) {
            int rowG = bm0 + wm*16 + vv + 8*g2;
            if (rowG < M) {
                long long ci = (long long)rowG * ldc + colG;
                float x = acc[vv] + bv;
                if (flags & GF_ACC)  x += C[ci];
                if (flags & GF_RELU) x = fmaxf(x, 0.f);
                C[ci] = x;
                if ((flags & GF_F16) && Ch) Ch[ci] = (h16)x;
            }
        }
    }
}

// ---------------------------------------------------------------------------
// Helper / pack / pointwise kernels
// ---------------------------------------------------------------------------
__global__ void k_dct(float* dm)
{
    int i = blockIdx.x * 256 + threadIdx.x;
    if (i >= 400) return;
    int k = i / 20, x = i % 20;
    float w = (k == 0) ? sqrtf(1.0f/20.0f) : sqrtf(2.0f/20.0f);
    dm[i] = w * cosf(3.14159265358979323846f * ((float)x + 0.5f) * (float)k / 20.0f);
}

__global__ void k_cast_f16(const float* s, h16* d, int total)
{
    int i = blockIdx.x * 256 + threadIdx.x;
    if (i >= total) return;
    d[i] = (h16)s[i];
}

// (edge-kernel operand) dst[c*R + r] = src[r*C + c]
__global__ void k_transpose_f16(const float* s, h16* d, int R, int C, int total)
{
    int i = blockIdx.x * 256 + threadIdx.x;
    if (i >= total) return;
    int r = i / C, c = i % C;
    d[(long long)c * R + r] = (h16)s[i];
}

// BT[n*Kp + k] = (k<Ks) ? W[(n+nOff)*srcld + k] : 0   (W is (N,K) row-major)
__global__ void k_packW_rowpad(const float* s, h16* d, int Ks, int Kp, int srcld,
                               int nOff, int total)
{
    int idx = blockIdx.x * 256 + threadIdx.x;
    if (idx >= total) return;
    int k = idx % Kp, n = idx / Kp;
    d[idx] = (k < Ks) ? (h16)s[(long long)(n + nOff)*srcld + k] : (h16)0.f;
}

// BT[n*Kp + k] = (k<Ks) ? W[k*N + n] : 0   (W is (K,N) row-major, transpose)
__global__ void k_packT_pad(const float* s, h16* d, int Ks, int Kp, int N, int total)
{
    int idx = blockIdx.x * 256 + threadIdx.x;
    if (idx >= total) return;
    int k = idx % Kp, n = idx / Kp;
    d[idx] = (k < Ks) ? (h16)s[(long long)k*N + n] : (h16)0.f;
}

// batched transpose: dst[(l*N + n)*K + k] = src[(l*K + k)*N + n]
__global__ void k_packWT3(const float* s, h16* d, int K, int N, int total)
{
    int idx = blockIdx.x * 256 + threadIdx.x;
    if (idx >= total) return;
    int k = idx % K;
    int r = idx / K;
    int n = r % N, l = r / N;
    d[idx] = (h16)s[((long long)l*K + k)*N + n];
}

// per-tap transposed conv weights: dst[(t*O + o)*I + i] = w[(o*I+i)*T + t]
__global__ void k_pack_convwT(const float* w, h16* d, int O, int I, int T, int total)
{
    int idx = blockIdx.x * 256 + threadIdx.x;
    if (idx >= total) return;
    int i = idx % I;
    int r = idx / I;
    int o = r % O, t = r / O;
    d[idx] = (h16)w[((long long)o*I + i)*T + t];
}

// pad columns Cs->Cp with zeros, f32 -> f16
__global__ void k_pad_cols(const float* s, h16* d, int Cs, int Cp, int total)
{
    int idx = blockIdx.x * 256 + threadIdx.x;
    if (idx >= total) return;
    int c = idx % Cp, r = idx / Cp;
    d[idx] = (c < Cs) ? (h16)s[(long long)r*Cs + c] : (h16)0.f;
}

// token matrix [(b*L+t), 64] = src[b, t0+t, off+f]*1e-3 (cols 54..63 zero)
__global__ void k_pack_tokens(const float* src, h16* dst, int L, int t0, int off, int total)
{
    int idx = blockIdx.x * 256 + threadIdx.x;
    if (idx >= total) return;
    int f = idx % 64;
    int r = idx / 64;
    int t = r % L, b = r / L;
    dst[idx] = (f < INF)
        ? (h16)(src[((long long)b*INN + t0 + t)*108 + off + f] * 1e-3f)
        : (h16)0.f;
}

// Tk[(s*10+l), c] = k1[(b*40 + v + l), c]
__global__ void k_gather_tk(const h16* k1, h16* Tk, int total)
{
    int idx = blockIdx.x * 256 + threadIdx.x;
    if (idx >= total) return;
    int c = idx & 255;
    int r = idx >> 8;
    int l = r % KSW, s = r / KSW;
    int b = s / VNW, v = s % VNW;
    Tk[idx] = k1[((long long)(b*40 + v + l))*DDIM + c];
}

// DCT value tokens: Tv[(s*20+l), f] = sum_ch dm[l,ch]*src[b, v+ch, off+f]
__global__ void k_val_tokens(const float* src, const float* dm, h16* Tv, int off, int total)
{
    int idx = blockIdx.x * 256 + threadIdx.x;
    if (idx >= total) return;
    int f = idx % INF;
    int r = idx / INF;
    int l = r % CHN, s = r / CHN;
    int b = s / VNW, v = s % VNW;
    float a = 0.f;
    for (int ch = 0; ch < CHN; ++ch)
        a += dm[l*20 + ch] * src[((long long)b*INN + v + ch)*108 + off + f];
    Tv[idx] = (h16)a;
}

// tiny per-window multi-head softmax attention
template<int L, int DH, int E>
__global__ void k_attn(const h16* __restrict__ Q, const h16* __restrict__ K,
                       const h16* __restrict__ V, h16* __restrict__ O, int total)
{
    int idx = blockIdx.x * 256 + threadIdx.x;
    if (idx >= total) return;
    constexpr int H = E / DH;
    int s  = idx / (H*L);
    int r  = idx % (H*L);
    int h  = r / L;
    int l1 = r % L;
    const long long rowQ = ((long long)s*L + l1)*E + h*DH;
    float q[DH];
    #pragma unroll
    for (int d = 0; d < DH; ++d) q[d] = (float)Q[rowQ + d];
    const float scl = rsqrtf((float)DH);
    float lg[L];
    #pragma unroll
    for (int l2 = 0; l2 < L; ++l2) {
        const long long rowK = ((long long)s*L + l2)*E + h*DH;
        float a = 0.f;
        #pragma unroll
        for (int d = 0; d < DH; ++d) a += q[d] * (float)K[rowK + d];
        lg[l2] = a * scl;
    }
    float mx = lg[0];
    #pragma unroll
    for (int l2 = 1; l2 < L; ++l2) mx = fmaxf(mx, lg[l2]);
    float sum = 0.f;
    #pragma unroll
    for (int l2 = 0; l2 < L; ++l2) { lg[l2] = __expf(lg[l2] - mx); sum += lg[l2]; }
    float inv = 1.f / sum;
    float o[DH];
    #pragma unroll
    for (int d = 0; d < DH; ++d) o[d] = 0.f;
    #pragma unroll
    for (int l2 = 0; l2 < L; ++l2) {
        const long long rowV = ((long long)s*L + l2)*E + h*DH;
        float p = lg[l2] * inv;
        #pragma unroll
        for (int d = 0; d < DH; ++d) o[d] += p * (float)V[rowV + d];
    }
    #pragma unroll
    for (int d = 0; d < DH; ++d) O[rowQ + d] = (h16)o[d];
}

// residual + LayerNorm over E
__global__ void k_ln(const h16* xt, const float* o, const float* g, const float* b,
                     h16* out, int rows, int E)
{
    int r = blockIdx.x * 256 + threadIdx.x;
    if (r >= rows) return;
    long long base = (long long)r * E;
    float mu = 0.f, m2 = 0.f;
    for (int c = 0; c < E; ++c) { float z = (float)xt[base+c] + o[base+c]; mu += z; m2 += z*z; }
    mu /= (float)E;
    float var = m2/(float)E - mu*mu;
    float inv = rsqrtf(var + 1e-5f);
    for (int c = 0; c < E; ++c) {
        float z = (float)xt[base+c] + o[base+c];
        out[base+c] = (h16)(((z - mu)*inv)*g[c] + b[c]);
    }
}

// GCN bn+tanh (+resid)
__global__ void k_bn_tanh(const float* in, const float* gm, const float* bt,
                          const float* resid, float* out, h16* outh, int total, int N)
{
    int idx = blockIdx.x * 256 + threadIdx.x;
    if (idx >= total) return;
    int col  = idx % N;
    int node = (idx / N) % INF;
    int nf   = node * N + col;
    float v = tanhf(gm[nf] * 0.9999950000374997f * in[idx] + bt[nf]);
    if (resid) v += resid[idx];
    out[idx] = v;
    outh[idx] = (h16)v;
}

__global__ void k_score(const float* q, const h16* kup, float* score, int total)
{
    int idx = blockIdx.x * 256 + threadIdx.x;
    if (idx >= total) return;
    int b = idx / VNW;
    float a = 1e-15f;
    for (int c = 0; c < DDIM; ++c)
        a += q[(long long)b*DDIM + c] * (float)kup[(long long)idx*DDIM + c];
    score[idx] = a;
}

__global__ void k_ssum(const float* score, float* ssum, int total)
{
    int b = blockIdx.x * 256 + threadIdx.x;
    if (b >= total) return;
    float a = 0.f;
    for (int v = 0; v < VNW; ++v) a += score[b*VNW + v];
    ssum[b] = a;
}

__global__ void k_dct_inp(const float* src, const float* dm, float* x0, int off, int total)
{
    int idx = blockIdx.x * 256 + threadIdx.x;
    if (idx >= total) return;
    int l = idx % 20;
    int r = idx / 20;
    int f = r % INF, b = r / INF;
    float a = 0.f;
    for (int ch = 0; ch < CHN; ++ch) {
        int t = (ch < 10) ? (40 + ch) : 49;
        a += dm[l*20 + ch] * src[((long long)b*INN + t)*108 + off + f];
    }
    x0[((long long)b*INF + f)*40 + l] = a;
}

__global__ void k_dct_att(const float* score, const float* ssum, const h16* vup,
                          float* x0, int total)
{
    int idx = blockIdx.x * 256 + threadIdx.x;
    if (idx >= total) return;
    int l = idx % 20;
    int r = idx / 20;
    int f = r % INF, b = r / INF;
    float inv = 1.f / ssum[b];
    float a = 0.f;
    for (int v = 0; v < VNW; ++v) {
        long long tok = (long long)(b*VNW + v)*CHN + l;
        a += score[b*VNW + v] * inv * (float)vup[tok*INF + f];
    }
    x0[((long long)b*INF + f)*40 + 20 + l] = a;
}

__global__ void k_final(const float* c2, const float* x0, const float* dm,
                        float* outp, int br, int total)
{
    int idx = blockIdx.x * 256 + threadIdx.x;
    if (idx >= total) return;
    int f = idx % INF;
    int r = idx / INF;
    int t = r % 20, b = r / 20;
    float a = 0.f;
    for (int l = 0; l < 20; ++l) {
        long long p = ((long long)b*INF + f)*40 + l;
        a += dm[l*20 + t] * (c2[p] + x0[p]);
    }
    outp[((long long)b*20 + t)*108 + br*54 + f] = a;
}

// ---------------------------------------------------------------------------
// Host orchestration
// ---------------------------------------------------------------------------
static void GEMMF(hipStream_t st, const h16* A, const h16* BT, float* C, h16* Ch,
                  const float* bias, int M, int N, int K, int lda, int ldb, int ldc,
                  int rD, int rM, int rA, int flags)
{
    dim3 grid((unsigned)(N/128), (unsigned)(M/64), 1);
    gemm_wmma_fast<<<grid, dim3(256), 0, st>>>(A, BT, C, Ch, bias,
                                               M, N, K, lda, ldb, ldc,
                                               rD, rM, rA, flags);
}

static void GEMME(hipStream_t st, const h16* A, const h16* B, float* C, h16* Ch,
                  const float* bias, int M, int N, int K, int lda, int ldb, int ldc,
                  long long sA, long long sB, long long sC, int batch,
                  int rD, int rM, int rA, int flags)
{
    dim3 grid((unsigned)((N + 63)/64), (unsigned)((M + 31)/32), (unsigned)batch);
    gemm_wmma<<<grid, dim3(256), 0, st>>>(A, B, C, Ch, bias, M, N, K, lda, ldb, ldc,
                                          sA, sB, sC, rD, rM, rA, flags);
}

#define LAUNCH(kern, total, ...) \
    kern<<<dim3((unsigned)(((total) + 255)/256)), dim3(256), 0, stream>>>(__VA_ARGS__)

extern "C" void kernel_launch(void* const* d_in, const int* in_sizes, int n_in,
                              void* d_out, int out_size, void* d_ws, size_t ws_size,
                              hipStream_t stream)
{
    (void)in_sizes; (void)n_in; (void)out_size; (void)ws_size;
    const float* src = (const float*)d_in[0];
    // params: 31 leaves per branch, b1 at d_in[1..31], b2 at d_in[32..62]
    auto PB = [&](int b, int k) -> const float* { return (const float*)d_in[1 + b*31 + k]; };

    size_t off = 0;
    char* base = (char*)d_ws;
    auto alloc = [&](size_t bytes) -> void* {
        void* p = base + off;
        off = (off + bytes + 255) & ~(size_t)255;
        return p;
    };

    float* dm = (float*)alloc(400 * 4);
    h16 *k1h[2], *q1h[2], *Tk[2], *Tv[2], *kup[2], *vup[2];
    float* qf[2];
    for (int b = 0; b < 2; ++b) {
        k1h[b] = (h16*)alloc((size_t)10240*256*2);
        q1h[b] = (h16*)alloc((size_t)2560*256*2);
        qf[b]  = (float*)alloc((size_t)256*256*4);
        Tk[b]  = (h16*)alloc((size_t)NSAMP*KSW*256*2);
        Tv[b]  = (h16*)alloc((size_t)NSAMP*CHN*54*2);
        kup[b] = (h16*)alloc((size_t)NSAMP*256*2);
        vup[b] = (h16*)alloc((size_t)NSAMP*CHN*54*2);
    }
    float* Cbig = (float*)alloc((size_t)79360*256*4);
    float* C2   = (float*)alloc((size_t)39680*256*4);
    h16* Qh  = (h16*)alloc((size_t)79360*256*2);
    h16* Kh  = (h16*)alloc((size_t)79360*256*2);
    h16* Vh  = (h16*)alloc((size_t)79360*256*2);
    h16* Oh  = (h16*)alloc((size_t)79360*256*2);
    h16* LNh = (h16*)alloc((size_t)79360*256*2);
    h16* K2h = (h16*)alloc((size_t)39680*256*2);
    h16* q2h = (h16*)alloc((size_t)1280*256*2);
    h16* Aq1 = (h16*)alloc((size_t)2560*64*2);
    h16* Ak1 = (h16*)alloc((size_t)10240*64*2);
    float* score = (float*)alloc((size_t)NSAMP*4);
    float* ssum  = (float*)alloc((size_t)BSZ*4);
    float* x0f = (float*)alloc((size_t)13824*40*4);
    h16*   x0h = (h16*)alloc((size_t)13824*64*2);
    float* Ya  = (float*)alloc((size_t)13824*256*4);
    h16*   Yah = (h16*)alloc((size_t)13824*256*2);
    float* Yb  = (float*)alloc((size_t)13824*256*4);
    h16*   Ybh = (h16*)alloc((size_t)13824*256*2);
    h16*   Hh  = (h16*)alloc((size_t)13824*256*2);
    // weight packs (f16). Fast-path packs are BT (n-major, k-contiguous);
    // edge-path packs keep [k][n].
    h16* Bqw1   = (h16*)alloc((size_t)256*64*2);
    h16* qw2p   = (h16*)alloc((size_t)6*256*256*2);
    h16* qw3p   = (h16*)alloc((size_t)5*256*256*2);
    h16* Bk1w   = (h16*)alloc((size_t)256*64*2);
    h16* k2w1p  = (h16*)alloc((size_t)6*256*256*2);
    h16* k2w2p  = (h16*)alloc((size_t)5*256*256*2);
    h16* xkinT  = (h16*)alloc((size_t)768*256*2);
    h16* xkoutT = (h16*)alloc((size_t)256*256*2);
    h16* xvinT  = (h16*)alloc((size_t)162*54*2);
    h16* xvoutT = (h16*)alloc((size_t)54*54*2);
    h16* ginw   = (h16*)alloc((size_t)256*64*2);
    h16* blkw   = (h16*)alloc((size_t)24*256*256*2);
    h16* goutw  = (h16*)alloc((size_t)256*40*2);
    h16* ginatt = (h16*)alloc((size_t)54*54*2);
    h16* blkatt = (h16*)alloc((size_t)24*54*54*2);
    h16* goutatt= (h16*)alloc((size_t)54*54*2);

    k_dct<<<dim3(2), dim3(256), 0, stream>>>(dm);

    // -------- Phase A: per-branch conv stacks + token/window materialization
    for (int b = 0; b < 2; ++b) {
        int offc = b * 54;
        LAUNCH(k_pack_tokens, 2560*64,  src, Aq1, KSW, 40, offc, 2560*64);
        LAUNCH(k_pack_tokens, 10240*64, src, Ak1, 40,  0,  offc, 10240*64);
        // qw1/k1w (256,54) -> BT[256][64] (rows n, k padded 54->64)
        LAUNCH(k_packW_rowpad, 256*64, PB(b,0), Bqw1, 54, 64, 54, 0, 256*64);
        LAUNCH(k_packW_rowpad, 256*64, PB(b,3), Bk1w, 54, 64, 54, 0, 256*64);
        // conv weights -> per-tap BT[o][i]
        LAUNCH(k_pack_convwT, 6*256*256, PB(b,1), qw2p, 256, 256, 6, 6*256*256);
        LAUNCH(k_pack_convwT, 5*256*256, PB(b,2), qw3p, 256, 256, 5, 5*256*256);
        // q1 = relu(squery/1000 @ qw1^T), k1 = relu(skey/1000 @ k1w^T)
        GEMMF(stream, Aq1, Bqw1, Cbig, q1h[b], nullptr, 2560, 256, 64, 64, 64, 256,
              0,0,0, GF_RELU|GF_F16);
        GEMMF(stream, Ak1, Bk1w, Cbig, k1h[b], nullptr, 10240, 256, 64, 64, 64, 256,
              0,0,0, GF_RELU|GF_F16);
        // q2 = relu(conv6(q1)) via tap-accumulated GEMMs
        for (int t = 0; t < 6; ++t)
            GEMMF(stream, q1h[b], qw2p + (size_t)t*65536, C2, q2h, nullptr,
                  1280, 256, 256, 256, 256, 256, 5, 10, t,
                  (t ? GF_ACC : 0) | (t == 5 ? (GF_RELU|GF_F16) : 0));
        // q3 = relu(conv5(q2)) -> qf[b]
        for (int t = 0; t < 5; ++t)
            GEMMF(stream, q2h, qw3p + (size_t)t*65536, qf[b], nullptr, nullptr,
                  256, 256, 256, 256, 256, 256, 1, 5, t,
                  (t ? GF_ACC : 0) | (t == 4 ? GF_RELU : 0));
        LAUNCH(k_gather_tk, NSAMP*KSW*256, k1h[b], Tk[b], NSAMP*KSW*256);
        LAUNCH(k_val_tokens, NSAMP*CHN*54, src, dm, Tv[b], offc, NSAMP*CHN*54);
    }

    // -------- Phase B: cross-branch xia attention + k2 convs
    for (int i = 0; i < 2; ++i) {
        int j = 1 - i;
        // k-side xia (E=256, L=10, H=8). in_w (768,256) is already n-major -> cast.
        LAUNCH(k_cast_f16, 768*256, PB(i,6), xkinT, 768*256);
        LAUNCH(k_cast_f16, 256*256, PB(i,8), xkoutT, 256*256);
        const float* inb = PB(i,7);
        GEMMF(stream, Tk[i], xkinT,            Cbig, Qh, inb,     79360, 256, 256, 256, 256, 256, 0,0,0, GF_BIAS|GF_F16);
        GEMMF(stream, Tk[j], xkinT + 256*256,  Cbig, Kh, inb+256, 79360, 256, 256, 256, 256, 256, 0,0,0, GF_BIAS|GF_F16);
        GEMMF(stream, Tk[j], xkinT + 512*256,  Cbig, Vh, inb+512, 79360, 256, 256, 256, 256, 256, 0,0,0, GF_BIAS|GF_F16);
        k_attn<10,32,256><<<dim3((NSAMP*8*10 + 255)/256), dim3(256), 0, stream>>>(Qh, Kh, Vh, Oh, NSAMP*8*10);
        GEMMF(stream, Oh, xkoutT, Cbig, nullptr, PB(i,9), 79360, 256, 256, 256, 256, 256, 0,0,0, GF_BIAS);
        LAUNCH(k_ln, 79360, Tk[i], Cbig, PB(i,10), PB(i,11), LNh, 79360, 256);
        // k2 conv stack
        LAUNCH(k_pack_convwT, 6*256*256, PB(i,4), k2w1p, 256, 256, 6, 6*256*256);
        LAUNCH(k_pack_convwT, 5*256*256, PB(i,5), k2w2p, 256, 256, 5, 5*256*256);
        for (int t = 0; t < 6; ++t)
            GEMMF(stream, LNh, k2w1p + (size_t)t*65536, Cbig, K2h, nullptr,
                  39680, 256, 256, 256, 256, 256, 5, 10, t,
                  (t ? GF_ACC : 0) | (t == 5 ? (GF_RELU|GF_F16) : 0));
        for (int t = 0; t < 5; ++t)
            GEMMF(stream, K2h, k2w2p + (size_t)t*65536, C2, kup[i], nullptr,
                  7936, 256, 256, 256, 256, 256, 1, 5, t,
                  (t ? GF_ACC : 0) | (t == 4 ? (GF_RELU|GF_F16) : 0));
        // v-side xia (E=54, L=20, H=6) -- ragged, general kernel ([k][n] packs)
        LAUNCH(k_transpose_f16, 162*54, PB(i,12), xvinT, 162, 54, 162*54);
        LAUNCH(k_transpose_f16, 54*54,  PB(i,14), xvoutT, 54, 54, 54*54);
        const float* inbv = PB(i,13);
        GEMME(stream, Tv[i], xvinT,     Cbig, Qh, inbv,     158720, 54, 54, 54, 162, 54, 0,0,0, 1, 0,0,0, GF_BIAS|GF_F16);
        GEMME(stream, Tv[j], xvinT+54,  Cbig, Kh, inbv+54,  158720, 54, 54, 54, 162, 54, 0,0,0, 1, 0,0,0, GF_BIAS|GF_F16);
        GEMME(stream, Tv[j], xvinT+108, Cbig, Vh, inbv+108, 158720, 54, 54, 54, 162, 54, 0,0,0, 1, 0,0,0, GF_BIAS|GF_F16);
        k_attn<20,9,54><<<dim3((NSAMP*6*20 + 255)/256), dim3(256), 0, stream>>>(Qh, Kh, Vh, Oh, NSAMP*6*20);
        GEMME(stream, Oh, xvoutT, Cbig, nullptr, PB(i,15), 158720, 54, 54, 54, 54, 54, 0,0,0, 1, 0,0,0, GF_BIAS);
        LAUNCH(k_ln, 158720, Tv[i], Cbig, PB(i,16), PB(i,17), vup[i], 158720, 54);
    }

    // -------- Phase C: score/att, DCT input, GCN, IDCT output
    for (int i = 0; i < 2; ++i) {
        int offc = i * 54;
        LAUNCH(k_score, NSAMP, qf[i], kup[i], score, NSAMP);
        LAUNCH(k_ssum, BSZ, score, ssum, BSZ);
        LAUNCH(k_dct_inp, BSZ*54*20, src, dm, x0f, offc, BSZ*54*20);
        LAUNCH(k_dct_att, BSZ*54*20, score, ssum, vup[i], x0f, BSZ*54*20);
        LAUNCH(k_pad_cols, 13824*64, x0f, x0h, 40, 64, 13824*64);
        // GCN weight packs: gin_w (40,256) -> BT[256][64]; blk_w -> BT per layer
        LAUNCH(k_packT_pad, 256*64,    PB(i,18), ginw, 40, 64, 256, 256*64);
        LAUNCH(k_packWT3, 24*256*256,  PB(i,23), blkw, 256, 256, 24*256*256);
        LAUNCH(k_cast_f16, 54*54,      PB(i,19), ginatt,  54*54);
        LAUNCH(k_cast_f16, 24*54*54,   PB(i,24), blkatt,  24*54*54);
        LAUNCH(k_cast_f16, 256*40,     PB(i,28), goutw,   256*40);
        LAUNCH(k_cast_f16, 54*54,      PB(i,29), goutatt, 54*54);
        // input layer: H = x0 @ gin_w  [K padded to 64]
        GEMMF(stream, x0h, ginw, Cbig, Hh, nullptr, 13824, 256, 64, 64, 64, 256, 0,0,0, GF_F16);
        GEMME(stream, ginatt, Hh, C2, nullptr, PB(i,20), 54, 256, 54, 54, 256, 256,
              0, 54*256, 54*256, BSZ, 0,0,0, GF_BIAS);
        LAUNCH(k_bn_tanh, 13824*256, C2, PB(i,21), PB(i,22), nullptr, Ya, Yah, 13824*256, 256);
        for (int s = 0; s < NSTG; ++s) {
            int l0 = s*2, l1 = s*2 + 1;
            GEMMF(stream, Yah, blkw + (size_t)l0*65536, Cbig, Hh, nullptr,
                  13824, 256, 256, 256, 256, 256, 0,0,0, GF_F16);
            GEMME(stream, blkatt + (size_t)l0*2916, Hh, C2, nullptr, PB(i,25) + (size_t)l0*256,
                  54, 256, 54, 54, 256, 256, 0, 54*256, 54*256, BSZ, 0,0,0, GF_BIAS);
            LAUNCH(k_bn_tanh, 13824*256, C2, PB(i,26) + (size_t)l0*13824, PB(i,27) + (size_t)l0*13824,
                   nullptr, Yb, Ybh, 13824*256, 256);
            GEMMF(stream, Ybh, blkw + (size_t)l1*65536, Cbig, Hh, nullptr,
                  13824, 256, 256, 256, 256, 256, 0,0,0, GF_F16);
            GEMME(stream, blkatt + (size_t)l1*2916, Hh, C2, nullptr, PB(i,25) + (size_t)l1*256,
                  54, 256, 54, 54, 256, 256, 0, 54*256, 54*256, BSZ, 0,0,0, GF_BIAS);
            LAUNCH(k_bn_tanh, 13824*256, C2, PB(i,26) + (size_t)l1*13824, PB(i,27) + (size_t)l1*13824,
                   Ya, Ya, Yah, 13824*256, 256);
        }
        // output layer + residual + IDCT
        GEMME(stream, Yah, goutw, Cbig, Hh, nullptr, 13824, 40, 256, 256, 40, 40,
              0,0,0, 1, 0,0,0, GF_F16);
        GEMME(stream, goutatt, Hh, C2, nullptr, PB(i,30), 54, 40, 54, 54, 40, 40,
              0, 54*40, 54*40, BSZ, 0,0,0, GF_BIAS);
        LAUNCH(k_final, BSZ*20*54, C2, x0f, dm, (float*)d_out, i, BSZ*20*54);
    }
}
